// Msa_34291018891646
// MI455X (gfx1250) — compile-verified, matrix operand order checked by reading, not run on hardware
//
#include <hip/hip_runtime.h>
#include <hip/hip_bf16.h>

// ---------------------------------------------------------------------------
// MHSA for MI455X (gfx1250): f16 WMMA (f32 accum), flash attention (64-key
// blocks), double-buffered LDS GEMMs, vectorized b128 staging, grouped
// fragment preloads for load/WMMA overlap.
// ---------------------------------------------------------------------------

typedef _Float16 h16;
typedef __attribute__((ext_vector_type(16))) _Float16 v16h;
typedef __attribute__((ext_vector_type(8)))  float    v8f;

static constexpr int Bdim = 4;
static constexpr int Nseq = 2048;
static constexpr int Cdim = 1024;
static constexpr int Hn   = 16;
static constexpr int HD   = 64;
static constexpr float SCALE = 0.125f;                         // 64^-0.5
static constexpr size_t BHND = (size_t)Bdim * Hn * Nseq * HD;  // 8388608

union Frag16 { v16h v; uint4 q[2]; };
union H4     { uint2 u; h16 hh[4]; };
union H8     { uint4 q; h16 hh[8]; };

// Load one 16x32 f16 fragment (A layout per CDNA5 ISA 7.12.2):
// lane L (L<16) row = L, K = {0..7, 16..23}; lane L+16 row = L, K = {8..15, 24..31}.
// tile points at fragment origin; stride in h16 elements; rows 16B aligned.
__device__ __forceinline__ v16h load_frag(const h16* tile, int stride, int lane) {
    const h16* p = tile + (size_t)(lane & 15) * stride + ((lane >> 4) << 3);
    Frag16 f;
    f.q[0] = *(const uint4*)(p);
    f.q[1] = *(const uint4*)(p + 16);
    return f.v;
}

__device__ __forceinline__ v8f wmma_f16(v16h a, v16h b, v8f c) {
    return __builtin_amdgcn_wmma_f32_16x16x32_f16(false, a, false, b,
                                                  (short)0, c, false, false);
}

__device__ __forceinline__ float half_max(float v) {
    #pragma unroll
    for (int m = 1; m < 16; m <<= 1) v = fmaxf(v, __shfl_xor(v, m, 32));
    return v;
}
__device__ __forceinline__ float half_sum(float v) {
    #pragma unroll
    for (int m = 1; m < 16; m <<= 1) v += __shfl_xor(v, m, 32);
    return v;
}

// ---------------------------------------------------------------------------
// Tiled GEMM: C[M,Nout] = A[M,K] * Bw[K,Nout] + bias.  Block tile 128x128,
// K-step 32, LDS double-buffered (ping-pong, one barrier/step), 8 waves each
// computing 32x64.  MODE 0: QKV epilogue (scatter f16 q/k/v).  MODE 1: proj
// epilogue (f32 out).
// ---------------------------------------------------------------------------
template <int MODE, typename TA>
__global__ __launch_bounds__(256)
void gemm_kernel(const TA* __restrict__ A, const float* __restrict__ Bw,
                 const float* __restrict__ bias, void* __restrict__ Out,
                 int M, int K, int Nout) {
    __shared__ __align__(16) h16 As[2][128 * 40];   // [row][k], stride 40
    __shared__ __align__(16) h16 Bs[2][128 * 40];   // transposed: [n][k]

    const int tid  = threadIdx.x;
    const int wave = tid >> 5;
    const int lane = tid & 31;
    const int m0   = blockIdx.y * 128;
    const int n0   = blockIdx.x * 128;
    const int wm   = (wave >> 1) * 32;   // 4 row groups of 32
    const int wn   = (wave & 1) * 64;    // 2 col groups of 64

    // Stage A tile 128x32 into buffer `buf` (b128 loads, b64 LDS stores)
    auto stageA = [&](int k0, int buf) {
        #pragma unroll
        for (int i = 0; i < 4; ++i) {
            int e = tid + 256 * i;          // 1024 slots of 4 elements
            int r = e >> 3, c = (e & 7) * 4;
            H4 h4;
            if constexpr (sizeof(TA) == 4) {
                float4 v4 = *(const float4*)&A[(size_t)(m0 + r) * K + k0 + c];
                h4.hh[0] = (h16)v4.x; h4.hh[1] = (h16)v4.y;
                h4.hh[2] = (h16)v4.z; h4.hh[3] = (h16)v4.w;
            } else {
                h4.u = *(const uint2*)&A[(size_t)(m0 + r) * K + k0 + c];
            }
            *(uint2*)&As[buf][r * 40 + c] = h4.u;
        }
    };
    // Stage B tile 32x128 transposed into Bs[buf][n][k]
    auto stageB = [&](int k0, int buf) {
        #pragma unroll
        for (int i = 0; i < 4; ++i) {
            int e = tid + 256 * i;
            int kr = e >> 5, nc = (e & 31) * 4;
            float4 v4 = *(const float4*)&Bw[(size_t)(k0 + kr) * Nout + n0 + nc];
            Bs[buf][(nc + 0) * 40 + kr] = (h16)v4.x;
            Bs[buf][(nc + 1) * 40 + kr] = (h16)v4.y;
            Bs[buf][(nc + 2) * 40 + kr] = (h16)v4.z;
            Bs[buf][(nc + 3) * 40 + kr] = (h16)v4.w;
        }
    };

    const v8f vzero = {0.f, 0.f, 0.f, 0.f, 0.f, 0.f, 0.f, 0.f};
    v8f acc[2][4];
    #pragma unroll
    for (int i = 0; i < 2; ++i)
        #pragma unroll
        for (int j = 0; j < 4; ++j) acc[i][j] = vzero;

    stageA(0, 0);
    stageB(0, 0);
    __syncthreads();

    const int steps = K / 32;
    for (int s = 0; s < steps; ++s) {
        const int buf = s & 1;
        if (s + 1 < steps) {            // overlap next tile's loads with WMMA
            stageA((s + 1) * 32, buf ^ 1);
            stageB((s + 1) * 32, buf ^ 1);
        }
        v16h af[2], bf[4];
        #pragma unroll
        for (int mi = 0; mi < 2; ++mi)
            af[mi] = load_frag(&As[buf][(size_t)(wm + 16 * mi) * 40], 40, lane);
        #pragma unroll
        for (int nj = 0; nj < 4; ++nj)
            bf[nj] = load_frag(&Bs[buf][(size_t)(wn + 16 * nj) * 40], 40, lane);
        #pragma unroll
        for (int mi = 0; mi < 2; ++mi)
            #pragma unroll
            for (int nj = 0; nj < 4; ++nj)
                acc[mi][nj] = wmma_f16(af[mi], bf[nj], acc[mi][nj]);
        __syncthreads();
    }

    // Epilogue.  D layout: reg r holds row (r + 8*(lane>>4)), col (lane&15).
    const int half = lane >> 4;
    #pragma unroll
    for (int mi = 0; mi < 2; ++mi) {
        #pragma unroll
        for (int nj = 0; nj < 4; ++nj) {
            #pragma unroll
            for (int r = 0; r < 8; ++r) {
                int row = m0 + wm + 16 * mi + r + 8 * half;
                int col = n0 + wn + 16 * nj + (lane & 15);
                float val = acc[mi][nj][r] + bias[col];
                if constexpr (MODE == 0) {
                    int which = col >> 10;
                    int rem   = col & 1023;
                    int h     = rem >> 6;
                    int d     = rem & 63;
                    int b     = row >> 11;
                    int nn    = row & 2047;
                    h16* q = (h16*)Out;
                    q[(size_t)which * BHND +
                      (((size_t)b * Hn + h) * Nseq + nn) * HD + d] = (h16)val;
                } else {
                    ((float*)Out)[(size_t)row * Nout + col] = val;
                }
            }
        }
    }
}

// ---------------------------------------------------------------------------
// Flash attention: block = (b*H+h, 128 query rows), 8 waves x 16 rows, Q held
// in registers.  64 keys per inner step: 8 QK^T WMMAs with K fragments
// preloaded in groups of 4 (clustered b128 loads -> staggered s_wait_loadcnt),
// one online-softmax update, 8 PV WMMAs (P via wave-private LDS, V staged
// transposed in LDS).
// ---------------------------------------------------------------------------
__global__ __launch_bounds__(256)
void attn_kernel(const h16* __restrict__ qkv, h16* __restrict__ attn_out) {
    __shared__ __align__(16) h16 Vt[64 * 72];        // [d][key], stride 72
    __shared__ __align__(16) h16 Pbuf[8 * 16 * 72];  // per-wave P [16][64+pad]

    const int tid  = threadIdx.x;
    const int wave = tid >> 5;
    const int lane = tid & 31;
    const int half = lane >> 4;
    const int bh   = blockIdx.y;          // b*H + h
    const int b    = bh >> 4;
    const int h    = bh & 15;
    const int q0   = blockIdx.x * 128;

    const h16* Q  = qkv + (size_t)bh * Nseq * HD;
    const h16* Kt = qkv + BHND + (size_t)bh * Nseq * HD;
    const h16* V  = qkv + 2 * BHND + (size_t)bh * Nseq * HD;
    h16* Pw = Pbuf + (size_t)wave * 16 * 72;

    const int qrow = q0 + wave * 16;
    const v16h qa0 = load_frag(Q + (size_t)qrow * HD,      HD, lane);  // d 0..31
    const v16h qa1 = load_frag(Q + (size_t)qrow * HD + 32, HD, lane);  // d 32..63

    const v8f vzero = {0.f, 0.f, 0.f, 0.f, 0.f, 0.f, 0.f, 0.f};
    v8f o[4];
    #pragma unroll
    for (int j = 0; j < 4; ++j) o[j] = vzero;
    float mrow[8], lrow[8];
    #pragma unroll
    for (int r = 0; r < 8; ++r) { mrow[r] = -1e30f; lrow[r] = 0.f; }

    for (int j0 = 0; j0 < Nseq; j0 += 64) {
        // Stage V[j0..j0+64) transposed -> Vt[d][key] (b128 loads)
        #pragma unroll
        for (int i = 0; i < 2; ++i) {
            int e = tid + 256 * i;          // 512 slots of 8 elements
            int kr = e >> 3, dc = (e & 7) * 8;
            H8 u;
            u.q = *(const uint4*)&V[(size_t)(j0 + kr) * HD + dc];
            #pragma unroll
            for (int t = 0; t < 8; ++t) Vt[(dc + t) * 72 + kr] = u.hh[t];
        }
        __syncthreads();

        if (j0 + 64 < Nseq) {               // prefetch next K/V block
            __builtin_prefetch((const void*)(Kt + (size_t)(j0 + 64) * HD), 0, 1);
            __builtin_prefetch((const void*)(V  + (size_t)(j0 + 64) * HD), 0, 1);
        }

        // S = Q K^T for 64 keys: two groups of two 16-key subtiles.  All four
        // fragments of a group are loaded before any of its WMMAs issue, so
        // the b128 loads clause and the waits stagger across WMMA execution.
        v8f s[4];
        #pragma unroll
        for (int g = 0; g < 2; ++g) {
            v16h kf[4];
            #pragma unroll
            for (int t = 0; t < 2; ++t) {
                const h16* kb = Kt + (size_t)(j0 + 16 * (2 * g + t)) * HD;
                kf[2 * t]     = load_frag(kb,      HD, lane);
                kf[2 * t + 1] = load_frag(kb + 32, HD, lane);
            }
            #pragma unroll
            for (int t = 0; t < 2; ++t) {
                v8f sv = vzero;
                sv = wmma_f16(qa0, kf[2 * t],     sv);
                sv = wmma_f16(qa1, kf[2 * t + 1], sv);
                s[2 * g + t] = sv;
            }
        }

        // Online softmax per row (row = r + 8*half, spread over a 16-lane half)
        #pragma unroll
        for (int r = 0; r < 8; ++r) {
            float x0 = s[0][r] * SCALE;
            float x1 = s[1][r] * SCALE;
            float x2 = s[2][r] * SCALE;
            float x3 = s[3][r] * SCALE;
            float mx = half_max(fmaxf(fmaxf(x0, x1), fmaxf(x2, x3)));
            float mnew = fmaxf(mrow[r], mx);
            float corr = __expf(mrow[r] - mnew);
            float p0 = __expf(x0 - mnew);
            float p1 = __expf(x1 - mnew);
            float p2 = __expf(x2 - mnew);
            float p3 = __expf(x3 - mnew);
            float rs = half_sum((p0 + p1) + (p2 + p3));
            lrow[r] = lrow[r] * corr + rs;
            mrow[r] = mnew;
            #pragma unroll
            for (int j = 0; j < 4; ++j) o[j][r] *= corr;
            int pidx = (r + 8 * half) * 72 + (lane & 15);
            Pw[pidx]      = (h16)p0;
            Pw[pidx + 16] = (h16)p1;
            Pw[pidx + 32] = (h16)p2;
            Pw[pidx + 48] = (h16)p3;
        }
        // Wave-private LDS round-trip: make P stores visible to the frag loads
        asm volatile("s_wait_dscnt 0" ::: "memory");

        // O += P @ V.  Preload both P chunks and the first V fragments so the
        // ds_load_b128s cluster ahead of the WMMA chain.
        {
            v16h pa0 = load_frag(Pw,      72, lane);
            v16h pa1 = load_frag(Pw + 32, 72, lane);
            #pragma unroll
            for (int g = 0; g < 2; ++g) {
                v16h bv[4];
                #pragma unroll
                for (int t = 0; t < 2; ++t) {
                    const h16* vb = Vt + (size_t)(16 * (2 * g + t)) * 72;
                    bv[2 * t]     = load_frag(vb,      72, lane);
                    bv[2 * t + 1] = load_frag(vb + 32, 72, lane);
                }
                #pragma unroll
                for (int t = 0; t < 2; ++t) {
                    int j = 2 * g + t;
                    o[j] = wmma_f16(pa0, bv[2 * t],     o[j]);
                    o[j] = wmma_f16(pa1, bv[2 * t + 1], o[j]);
                }
            }
        }
        __syncthreads();   // protect Vt before next stage overwrites it
    }

    // Normalize and store f16 attn_out[b][nq][h*HD+d] (row-major [B*N][C])
    #pragma unroll
    for (int j = 0; j < 4; ++j) {
        #pragma unroll
        for (int r = 0; r < 8; ++r) {
            float val = o[j][r] / lrow[r];
            int nq = qrow + r + 8 * half;
            int c  = h * HD + 16 * j + (lane & 15);
            attn_out[((size_t)b * Nseq + nq) * Cdim + c] = (h16)val;
        }
    }
}

// ---------------------------------------------------------------------------
extern "C" void kernel_launch(void* const* d_in, const int* in_sizes, int n_in,
                              void* d_out, int out_size, void* d_ws, size_t ws_size,
                              hipStream_t stream) {
    const float* x      = (const float*)d_in[0];
    const float* w_qkv  = (const float*)d_in[1];
    const float* b_qkv  = (const float*)d_in[2];
    const float* w_proj = (const float*)d_in[3];
    const float* b_proj = (const float*)d_in[4];
    float* out = (float*)d_out;

    h16* qkv  = (h16*)d_ws;              // [3][B][H][N][HD] f16 (50.3 MB)
    h16* attn = qkv + 3 * BHND;          // [B*N][C] f16 (16.8 MB)

    const int M = Bdim * Nseq;           // 8192
    dim3 blk(256);

    // QKV: [8192,1024] x [1024,3072] -> scattered f16 q/k/v
    gemm_kernel<0, float><<<dim3(3 * Cdim / 128, M / 128), blk, 0, stream>>>(
        x, w_qkv, b_qkv, (void*)qkv, M, Cdim, 3 * Cdim);

    // Flash attention: grid (query tiles, B*H)
    attn_kernel<<<dim3(Nseq / 128, Bdim * Hn), blk, 0, stream>>>(qkv, attn);

    // Proj: [8192,1024] x [1024,1024] + bias -> f32 out
    gemm_kernel<1, h16><<<dim3(Cdim / 128, M / 128), blk, 0, stream>>>(
        attn, w_proj, b_proj, (void*)out, M, Cdim, Cdim);
}
